// BasicBlock_quant_20572893348555
// MI455X (gfx1250) — matrix operand order repaired and checked by reading, not compiled, not run on hardware
//
#include <hip/hip_runtime.h>
#include <hip/hip_bf16.h>

// ---------------------------------------------------------------------------
// BasicBlock (quantized) for MI455X / gfx1250:
//   conv3x3(wq8) + bias -> BN(train) -> ReLU -> conv3x3(wq8) + bias -> BN ->
//   +identity -> ReLU
// Core compute on v_wmma_f32_16x16x32_bf16 (wave32 WMMA), f32 accumulation.
// Halo staging uses GLOBAL_LOAD_ASYNC_TO_LDS (ASYNCcnt) when available.
// ---------------------------------------------------------------------------

typedef __attribute__((ext_vector_type(16))) __bf16 v16bf;
typedef __attribute__((ext_vector_type(8)))  __bf16 v8bf;
typedef __attribute__((ext_vector_type(8)))  float  v8f;

#define CCH 128          // channels (in == out)
#define HH  56
#define WW  56
#define NB  64           // batch
#define NPIX (NB*HH*WW)  // 200704 pixels
#define NTOT ((size_t)NPIX * CCH)

#if defined(__gfx1250__) && \
    __has_builtin(__builtin_amdgcn_global_load_async_to_lds_b128) && \
    __has_builtin(__builtin_amdgcn_s_wait_asynccnt)
#define USE_ASYNC_LDS 1
// builtin prototype (from diagnostics): (v4i addrspace(1)*, v4i addrspace(3)*, Imm, Imm)
typedef int async_v4i __attribute__((vector_size(16)));
#define ASYNC_CP16(dst_lds, src_glob)                                      \
    __builtin_amdgcn_global_load_async_to_lds_b128(                        \
        (__attribute__((address_space(1))) async_v4i*)(src_glob),          \
        (__attribute__((address_space(3))) async_v4i*)(dst_lds),           \
        /*imm offset*/0, /*cpol*/0)
#else
#define USE_ASYNC_LDS 0
#endif

// ---------------- weight quantization ---------------------------------------

__global__ void absmax_kernel(const float* __restrict__ w, unsigned int* __restrict__ outbits, int n) {
    __shared__ float red[256];
    float m = 0.f;
    for (int i = blockIdx.x * blockDim.x + threadIdx.x; i < n; i += gridDim.x * blockDim.x)
        m = fmaxf(m, fabsf(w[i]));
    red[threadIdx.x] = m;
    __syncthreads();
    for (int s = 128; s > 0; s >>= 1) {
        if ((int)threadIdx.x < s) red[threadIdx.x] = fmaxf(red[threadIdx.x], red[threadIdx.x + s]);
        __syncthreads();
    }
    if (threadIdx.x == 0) atomicMax(outbits, __float_as_uint(red[0]));  // vals >= 0: bit order == float order
}

// OIHW [K][C][3][3] f32  ->  quantized bf16 packed [rs][C][K]  (K contiguous)
__global__ void quant_pack_kernel(const float* __restrict__ w,
                                  const unsigned int* __restrict__ maxbits,
                                  __bf16* __restrict__ wq) {
    int idx = blockIdx.x * 256 + threadIdx.x;
    if (idx >= CCH * CCH * 9) return;
    int k   = idx / (CCH * 9);
    int rem = idx % (CCH * 9);
    int c   = rem / 9;
    int rs  = rem % 9;
    float scale = __uint_as_float(*maxbits) / 127.f;
    float v  = w[idx];
    float qv = rintf(v / scale) * scale;            // symmetric 8-bit levels: exact in bf16 mantissa
    wq[((size_t)rs * CCH + c) * CCH + k] = (__bf16)qv;
}

// ---------------- activation layout convert ---------------------------------

// NCHW f32 -> NHWC bf16
__global__ void nchw_to_nhwc_bf16_kernel(const float* __restrict__ x, __bf16* __restrict__ xh) {
    for (size_t i = (size_t)blockIdx.x * blockDim.x + threadIdx.x; i < NTOT;
         i += (size_t)gridDim.x * blockDim.x) {
        int c      = (int)(i & (CCH - 1));
        size_t pix = i >> 7;                 // n*3136 + h*56 + w
        size_t n   = pix / (HH * WW);
        size_t hw  = pix % (HH * WW);
        xh[i] = (__bf16)x[(n * CCH + c) * (HH * WW) + hw];
    }
}

// ---------------- implicit-GEMM 3x3 conv on WMMA ----------------------------
//
// Block: 256 threads (8 wave32), one (n, 8x8 pixel tile). Output tile 64x128.
// Waves in a 2(M) x 4(N) grid; each wave: 32x32 outputs = 2x2 wmma accums.
// LDS: 10x10 halo x 128 c in bf16 (25.6 KB), staged once (async-to-LDS when
// available), reused for all 9 taps x 4 c-chunks -> 144 v_wmma per wave.

__global__ __launch_bounds__(256) void conv3x3_wmma_kernel(
        const __bf16* __restrict__ xh,    // NHWC bf16
        const __bf16* __restrict__ wq,    // [9][C][K] bf16 (quantized)
        const float*  __restrict__ bias,  // [K]
        float*        __restrict__ y)     // NHWC f32
{
    __shared__ __bf16 lds[10 * 10 * CCH];     // 25600 B

    const int n   = blockIdx.y;
    const int th  = blockIdx.x / 7;
    const int tw  = blockIdx.x % 7;
    const int h0  = th * 8, w0 = tw * 8;
    const int tid = threadIdx.x;

    // ---- stage 10x10x128 bf16 halo (zero-padded) : 1600 x 16B chunks ----
#if USE_ASYNC_LDS
    for (int ch = tid; ch < 1600; ch += 256) {
        int p  = ch >> 4;            // halo pixel 0..99
        int cc = ch & 15;            // 8-element channel chunk
        int ph = p / 10, pw = p % 10;
        int h = h0 + ph - 1, w = w0 + pw - 1;
        __bf16* lp = &lds[p * CCH + cc * 8];
        if (h >= 0 && h < HH && w >= 0 && w < WW) {
            const __bf16* gp = xh + (((size_t)n * HH + h) * WW + w) * CCH + cc * 8;
            // DMA 16B global -> LDS without VGPR round-trip (tracked by ASYNCcnt)
            ASYNC_CP16(lp, gp);
        } else {
            v8bf zz = {};            // zero-pad chunks (disjoint from async targets)
            *(v8bf*)lp = zz;
        }
    }
    __builtin_amdgcn_s_wait_asynccnt(0);
#else
    for (int ch = tid; ch < 1600; ch += 256) {
        int p  = ch >> 4;
        int cc = ch & 15;
        int ph = p / 10, pw = p % 10;
        int h = h0 + ph - 1, w = w0 + pw - 1;
        v8bf v = {};
        if (h >= 0 && h < HH && w >= 0 && w < WW)
            v = *(const v8bf*)(xh + (((size_t)n * HH + h) * WW + w) * CCH + cc * 8);
        *(v8bf*)(&lds[p * CCH + cc * 8]) = v;
    }
#endif
    __syncthreads();

    const int wave  = tid >> 5;
    const int lane  = tid & 31;
    const int wm    = wave >> 2;     // 0..1 : M group of 32
    const int wn    = wave & 3;      // 0..3 : N group of 32
    const int lhalf = lane >> 4;     // 0/1
    const int l15   = lane & 15;

    v8f acc[2][2] = {};

    for (int rs = 0; rs < 9; ++rs) {
        const int r = rs / 3, s = rs % 3;
        const __bf16* wrs = wq + (size_t)rs * CCH * CCH;
        if (rs < 8)  // pull next tap's weight panel toward L2/L0 (global_prefetch_b8)
            __builtin_prefetch(wrs + CCH * CCH, 0, 3);
        for (int kc = 0; kc < 4; ++kc) {
            const int c0 = kc * 32;

            // B fragments (32x16 bf16): lane holds column kbase+l15, elems over c in A's table order
            v16bf bfrag[2];
#pragma unroll
            for (int j = 0; j < 2; ++j) {
                const int kbase = wn * 32 + j * 16;
#pragma unroll
                for (int e = 0; e < 16; ++e) {
                    const int kk = (e < 8) ? (lhalf * 8 + e) : (16 + lhalf * 8 + (e - 8));
                    bfrag[j][e] = wrs[(size_t)(c0 + kk) * CCH + kbase + l15];
                }
            }

#pragma unroll
            for (int i = 0; i < 2; ++i) {
                // A fragment (16x32 bf16), ISA layout:
                //   elems 0..7  -> K = c0 + lhalf*8 + 0..7
                //   elems 8..15 -> K = c0 + 16 + lhalf*8 + 0..7
                const int m  = wm * 32 + i * 16 + l15;   // local pixel 0..63
                const int ph = (m >> 3) + r;
                const int pw = (m & 7) + s;
                const __bf16* src = &lds[(ph * 10 + pw) * CCH + c0];
                union { v16bf v; v8bf h[2]; } a;
                a.h[0] = *(const v8bf*)(src + lhalf * 8);
                a.h[1] = *(const v8bf*)(src + 16 + lhalf * 8);
#pragma unroll
                for (int j = 0; j < 2; ++j) {
                    acc[i][j] = __builtin_amdgcn_wmma_f32_16x16x32_bf16(
                        false, a.v, false, bfrag[j], (short)0, acc[i][j], false, false);
                }
            }
        }
    }

    // ---- epilogue: + bias, store NHWC f32 ----
    // C/D layout: lanes 0-15 -> N=lane, M=q ; lanes 16-31 -> N=lane-16, M=q+8
#pragma unroll
    for (int i = 0; i < 2; ++i) {
#pragma unroll
        for (int j = 0; j < 2; ++j) {
            const int kg = wn * 32 + j * 16 + l15;
            const float bv = bias[kg];
#pragma unroll
            for (int q = 0; q < 8; ++q) {
                const int m = wm * 32 + i * 16 + q + lhalf * 8;
                const int h = h0 + (m >> 3), w = w0 + (m & 7);
                y[(((size_t)n * HH + h) * WW + w) * CCH + kg] = acc[i][j][q] + bv;
            }
        }
    }
}

// ---------------- BatchNorm (training-mode batch stats) ---------------------

// per-channel sum / sumsq over NHWC f32 tensor
__global__ void bn_stats_kernel(const float* __restrict__ y,
                                float* __restrict__ sum, float* __restrict__ sumsq) {
    __shared__ float s1[256], s2[256];
    const int c   = threadIdx.x & (CCH - 1);
    const int sub = threadIdx.x >> 7;          // 0/1
    float a = 0.f, b = 0.f;
    for (int p = blockIdx.x * 2 + sub; p < NPIX; p += gridDim.x * 2) {
        float v = y[(size_t)p * CCH + c];
        a += v;
        b += v * v;
    }
    s1[threadIdx.x] = a; s2[threadIdx.x] = b;
    __syncthreads();
    if (threadIdx.x < 128) {
        a = s1[threadIdx.x] + s1[threadIdx.x + 128];
        b = s2[threadIdx.x] + s2[threadIdx.x + 128];
        atomicAdd(&sum[c], a);
        atomicAdd(&sumsq[c], b);
    }
}

// BN -> ReLU -> bf16 (input for conv2), NHWC
__global__ void bn_relu_bf16_kernel(const float* __restrict__ y,
                                    const float* __restrict__ sum, const float* __restrict__ sumsq,
                                    const float* __restrict__ gamma, const float* __restrict__ beta,
                                    __bf16* __restrict__ out) {
    const float inv_m = 1.0f / (float)NPIX;
    for (size_t i = (size_t)blockIdx.x * blockDim.x + threadIdx.x; i < NTOT;
         i += (size_t)gridDim.x * blockDim.x) {
        int c = (int)(i & (CCH - 1));
        float mu  = sum[c] * inv_m;
        float var = sumsq[c] * inv_m - mu * mu;
        float rsd = rsqrtf(var + 1e-5f);
        float v   = (y[i] - mu) * rsd * gamma[c] + beta[c];
        out[i] = (__bf16)fmaxf(v, 0.f);
    }
}

// BN2 -> + identity (NCHW f32) -> ReLU -> out NCHW f32
__global__ void bn_add_relu_out_kernel(const float* __restrict__ z,  // NHWC f32
                                       const float* __restrict__ sum, const float* __restrict__ sumsq,
                                       const float* __restrict__ gamma, const float* __restrict__ beta,
                                       const float* __restrict__ x,  // NCHW f32 identity
                                       float* __restrict__ out) {    // NCHW f32
    const float inv_m = 1.0f / (float)NPIX;
    for (size_t i = (size_t)blockIdx.x * blockDim.x + threadIdx.x; i < NTOT;
         i += (size_t)gridDim.x * blockDim.x) {
        int c      = (int)(i & (CCH - 1));
        size_t pix = i >> 7;
        size_t n   = pix / (HH * WW);
        size_t hw  = pix % (HH * WW);
        size_t oi  = (n * CCH + c) * (HH * WW) + hw;
        float mu  = sum[c] * inv_m;
        float var = sumsq[c] * inv_m - mu * mu;
        float rsd = rsqrtf(var + 1e-5f);
        float v   = (z[i] - mu) * rsd * gamma[c] + beta[c] + x[oi];
        out[oi] = fmaxf(v, 0.f);
    }
}

// ---------------------------------------------------------------------------

extern "C" void kernel_launch(void* const* d_in, const int* in_sizes, int n_in,
                              void* d_out, int out_size, void* d_ws, size_t ws_size,
                              hipStream_t stream) {
    (void)in_sizes; (void)n_in; (void)out_size; (void)ws_size;

    const float* x      = (const float*)d_in[0];  // [64,128,56,56]
    const float* w1     = (const float*)d_in[1];  // [128,128,3,3]
    const float* b1     = (const float*)d_in[2];
    const float* gamma1 = (const float*)d_in[3];
    const float* beta1  = (const float*)d_in[4];
    const float* w2     = (const float*)d_in[5];
    const float* b2     = (const float*)d_in[6];
    const float* gamma2 = (const float*)d_in[7];
    const float* beta2  = (const float*)d_in[8];
    float* out = (float*)d_out;

    // ---- workspace carve-up (aliased; ~155 MB total) ----
    char* ws = (char*)d_ws;
    __bf16*       wq1   = (__bf16*)(ws + 0);                       // 294912 B
    __bf16*       wq2   = (__bf16*)(ws + 294912);                  // 294912 B
    float*        stats = (float*)(ws + 589824);                   // 512 f32: sum1,sq1,sum2,sq2
    unsigned int* maxw  = (unsigned int*)(ws + 589824 + 2048);     // 2 u32
    __bf16*       xh    = (__bf16*)(ws + (1 << 20));               // 51380224 B (NHWC bf16)
    float*        y     = (float*)(ws + (1 << 20) + 51380224);     // 102760448 B (NHWC f32)
    __bf16*       y2h   = xh;   // alias: xh dead after conv1
    float*        z     = y;    // alias: y dead after bn_relu

    float* sum1 = stats, *sq1 = stats + 128, *sum2 = stats + 256, *sq2 = stats + 384;

    // zero stats + absmax accumulators (graph-capturable memset node)
    (void)hipMemsetAsync(stats, 0, 4096, stream);

    const int NW = CCH * CCH * 9;   // 147456 weight elements

    // weight quantization (both convs)
    absmax_kernel<<<64, 256, 0, stream>>>(w1, &maxw[0], NW);
    absmax_kernel<<<64, 256, 0, stream>>>(w2, &maxw[1], NW);
    quant_pack_kernel<<<(NW + 255) / 256, 256, 0, stream>>>(w1, &maxw[0], wq1);
    quant_pack_kernel<<<(NW + 255) / 256, 256, 0, stream>>>(w2, &maxw[1], wq2);

    // activations -> NHWC bf16
    nchw_to_nhwc_bf16_kernel<<<4096, 256, 0, stream>>>(x, xh);

    // conv1 (WMMA implicit GEMM) -> y (NHWC f32)
    conv3x3_wmma_kernel<<<dim3(49, NB), 256, 0, stream>>>(xh, wq1, b1, y);

    // BN1 stats + apply + ReLU -> bf16 NHWC
    bn_stats_kernel<<<256, 256, 0, stream>>>(y, sum1, sq1);
    bn_relu_bf16_kernel<<<4096, 256, 0, stream>>>(y, sum1, sq1, gamma1, beta1, y2h);

    // conv2 -> z (NHWC f32)
    conv3x3_wmma_kernel<<<dim3(49, NB), 256, 0, stream>>>(y2h, wq2, b2, z);

    // BN2 stats + (normalize + residual + ReLU) -> NCHW f32 output
    bn_stats_kernel<<<256, 256, 0, stream>>>(z, sum2, sq2);
    bn_add_relu_out_kernel<<<4096, 256, 0, stream>>>(z, sum2, sq2, gamma2, beta2, x, out);
}
